// DF_RFEM_2_85048942395453
// MI455X (gfx1250) — matrix operand order repaired and checked by reading, not compiled, not run on hardware
//
#include <hip/hip_runtime.h>
#include <math.h>

#define BB 4
#define HH 64
#define WW 64
#define HW 4096           // 64*64
#define IN_CH 256
#define SKIP_CH 64
#define EMBED 64
#define QC 32

typedef __attribute__((ext_vector_type(16))) __bf16 v16bf;
typedef __attribute__((ext_vector_type(8)))  __bf16 v8bf;
typedef __attribute__((ext_vector_type(8)))  float  v8f;

// ---------------------------------------------------------------------------
// CDNA5 wave32 WMMA fragment layouts (CDNA5 ISA 7.12.2):
//   A 16x32 bf16: lane m=lane&15, half=lane>>4; elem e -> K = (e>>3)*16 + half*8 + (e&7)
//     => two contiguous 8-elem (16B) runs at K = half*8 and K = 16 + half*8
//   B 32x16 bf16: lane n=lane&15; elem e -> K = (e>>3)*16 + half*8 + (e&7) (same as A)
//     => two contiguous 8-elem (16B) runs at K = half*8 and K = 16 + half*8
//   C/D 16x16 f32: vgpr r -> M = r + 8*half (8 consecutive M per lane), N = lane&15
// ---------------------------------------------------------------------------
__device__ __forceinline__ float gelu_exact(float v) {
    return 0.5f * v * (1.0f + erff(v * 0.7071067811865476f));
}

__device__ __forceinline__ v16bf ld16(const __bf16* lo, const __bf16* hi) {
    const v8bf a = *(const v8bf*)lo;      // global_load_b128
    const v8bf b = *(const v8bf*)hi;      // global_load_b128
    v16bf r;
#pragma unroll
    for (int i = 0; i < 8; ++i) { r[i] = a[i]; r[i + 8] = b[i]; }
    return r;
}

__device__ __forceinline__ void st8bf(__bf16* p, const v8f v) {
    v8bf t;
#pragma unroll
    for (int r = 0; r < 8; ++r) t[r] = (__bf16)v[r];
    *(v8bf*)p = t;                        // global_store_b128
}

// core: acc += W[MxK](bf16,row-major) * Bt[N x K](bf16, pixel-major) for one 16x16 tile
__device__ __forceinline__ v8f bf_gemm(const __bf16* __restrict__ A, int lda,
                                       const __bf16* __restrict__ Bt, int ldb,
                                       int K, int m0, int n0) {
    const int lane = threadIdx.x & 31;
    const int half = lane >> 4, lm = lane & 15;
    const __bf16* ar = A + (size_t)(m0 + lm) * lda;
    const __bf16* br = Bt + (size_t)(n0 + lm) * ldb;
    v8f acc = {};
    for (int k0 = 0; k0 < K; k0 += 32) {
        const v16bf a = ld16(ar + k0 + half * 8, ar + k0 + 16 + half * 8);
        const v16bf b = ld16(br + k0 + half * 8, br + k0 + 16 + half * 8);
        acc = __builtin_amdgcn_wmma_f32_16x16x32_bf16(false, a, false, b,
                                                      (short)0, acc, false, false);
    }
    return acc;
}

// ---------------------------------------------------------------------------
// Templated GEMM over pixel-major activations.
//   RES: 0 none | 1 bf16 transposed [N][M] | 2 f32 transposed [N][M]
//   ACT: 1 -> exact GELU
//   OUT: 0 bf16 transposed [N][LDO] at column CO | 1 f32 transposed [N][LDO] | 2 f32 CHW
// ---------------------------------------------------------------------------
template <int M, int K, int LDB, int RES, int ACT, int OUT, int LDO, int CO>
__global__ __launch_bounds__(128)
void tgemm_kernel(const __bf16* __restrict__ W, const __bf16* __restrict__ Bt,
                  const float* __restrict__ bias, const void* __restrict__ res,
                  void* __restrict__ out) {
    const int wave = (blockIdx.x << 2) + (threadIdx.x >> 5);
    const int tiles = (M >> 4) << 8;     // (M/16) * 256 n-tiles
    const int b = wave / tiles, t = wave - b * tiles;
    const int m0 = (t >> 8) << 4, n0 = (t & 255) << 4;
    const int lane = threadIdx.x & 31, half = lane >> 4, lm = lane & 15;

    v8f acc = bf_gemm(W, K, Bt + (size_t)b * HW * LDB, LDB, K, m0, n0);

    const int mb_ = m0 + (half << 3);    // first of 8 consecutive output channels
    const size_t row = (size_t)(b * HW + n0 + lm);
    v8f v;
#pragma unroll
    for (int r = 0; r < 8; ++r) v[r] = acc[r] + bias[mb_ + r];
    if (RES == 1) {
        const v8bf rb = *(const v8bf*)((const __bf16*)res + row * M + mb_);
#pragma unroll
        for (int r = 0; r < 8; ++r) v[r] += (float)rb[r];
    } else if (RES == 2) {
        const float* rp = (const float*)res + row * M + mb_;
#pragma unroll
        for (int r = 0; r < 8; ++r) v[r] += rp[r];
    }
    if (ACT) {
#pragma unroll
        for (int r = 0; r < 8; ++r) v[r] = gelu_exact(v[r]);
    }
    if (OUT == 0) {
        st8bf((__bf16*)out + row * LDO + CO + mb_, v);
    } else if (OUT == 1) {
        float* op = (float*)out + row * LDO + mb_;
#pragma unroll
        for (int r = 0; r < 8; ++r) op[r] = v[r];
    } else {
        float* op = (float*)out + ((size_t)b * M + mb_) * HW + n0 + lm;
#pragma unroll
        for (int r = 0; r < 8; ++r) op[(size_t)r * HW] = v[r];
    }
}

// qkv GEMM: dual-layout output (q,k pixel-major for attention A/B frags; v channel-major)
__global__ __launch_bounds__(128)
void qkv_gemm_kernel(const __bf16* __restrict__ W, const __bf16* __restrict__ xT,
                     const float* __restrict__ bias, __bf16* __restrict__ qkT,
                     __bf16* __restrict__ vbf) {
    const int wave = (blockIdx.x << 2) + (threadIdx.x >> 5);
    const int tiles = 6 * 256;           // M=96
    const int b = wave / tiles, t = wave - b * tiles;
    const int m0 = (t >> 8) << 4, n0 = (t & 255) << 4;
    const int lane = threadIdx.x & 31, half = lane >> 4, lm = lane & 15;

    v8f acc = bf_gemm(W, IN_CH, xT + (size_t)b * HW * IN_CH, IN_CH, IN_CH, m0, n0);
    const int mb_ = m0 + (half << 3);
    v8f v;
#pragma unroll
    for (int r = 0; r < 8; ++r) v[r] = acc[r] + bias[mb_ + r];
    if (m0 < 64) {                       // q,k -> [pos][64]
        st8bf(qkT + (size_t)(b * HW + n0 + lm) * 64 + mb_, v);
    } else {                             // v -> [32][HW]
        __bf16* vb = vbf + (size_t)b * QC * HW;
#pragma unroll
        for (int r = 0; r < 8; ++r) vb[(size_t)(mb_ - 64 + r) * HW + n0 + lm] = (__bf16)v[r];
    }
}

// ---------------------------------------------------------------------------
// Flash attention: one 16-query tile per wave, keys streamed 32 at a time.
// All WMMA fragments are b128 vector loads; P relayout via LDS + s_wait_dscnt.
// ---------------------------------------------------------------------------
__global__ __launch_bounds__(128)
void attn_kernel(const __bf16* __restrict__ qkT, const __bf16* __restrict__ vbf,
                 __bf16* __restrict__ avT) {
    const int wave  = (blockIdx.x << 2) + (threadIdx.x >> 5);
    const int b     = wave >> 8;
    const int qbase = (wave & 255) << 4;
    const int lane  = threadIdx.x & 31;
    const int half  = lane >> 4, lm = lane & 15;
    const float scale = 0.17677669529663687f;   // 1/sqrt(32)

    const __bf16* qk = qkT + (size_t)b * HW * 64;
    const __bf16* vb = vbf + (size_t)b * QC * HW;

    __shared__ __bf16 lds_p[4 * 16 * 32];
    __bf16* pbuf = lds_p + ((threadIdx.x >> 5) << 9);

    const __bf16* qrow = qk + (size_t)(qbase + lm) * 64;            // channels 0..31
    const v16bf aq = ld16(qrow + half * 8, qrow + 16 + half * 8);

    float mrow[8], lrow[8];
    v8f acc0 = {}, acc1 = {};
#pragma unroll
    for (int r = 0; r < 8; ++r) { mrow[r] = -1e30f; lrow[r] = 0.0f; }

    for (int kt = 0; kt < HW; kt += 32) {
        const __bf16* kr0 = qk + (size_t)(kt + lm) * 64 + 32;       // channels 32..63
        const __bf16* kr1 = qk + (size_t)(kt + 16 + lm) * 64 + 32;
        const v16bf bk0 = ld16(kr0 + half * 8, kr0 + 16 + half * 8);
        const v16bf bk1 = ld16(kr1 + half * 8, kr1 + 16 + half * 8);
        v8f s0 = {}, s1 = {};
        s0 = __builtin_amdgcn_wmma_f32_16x16x32_bf16(false, aq, false, bk0, (short)0, s0, false, false);
        s1 = __builtin_amdgcn_wmma_f32_16x16x32_bf16(false, aq, false, bk1, (short)0, s1, false, false);

#pragma unroll
        for (int r = 0; r < 8; ++r) {
            const float sv0 = s0[r] * scale;
            const float sv1 = s1[r] * scale;
            float mx = fmaxf(sv0, sv1);
            mx = fmaxf(mx, __shfl_xor(mx, 1, 32));
            mx = fmaxf(mx, __shfl_xor(mx, 2, 32));
            mx = fmaxf(mx, __shfl_xor(mx, 4, 32));
            mx = fmaxf(mx, __shfl_xor(mx, 8, 32));
            const float newm = fmaxf(mrow[r], mx);
            const float corr = __expf(mrow[r] - newm);
            mrow[r] = newm;
            const float p0 = __expf(sv0 - newm);
            const float p1 = __expf(sv1 - newm);
            float rs = p0 + p1;
            rs += __shfl_xor(rs, 1, 32);
            rs += __shfl_xor(rs, 2, 32);
            rs += __shfl_xor(rs, 4, 32);
            rs += __shfl_xor(rs, 8, 32);
            lrow[r] = lrow[r] * corr + rs;
            acc0[r] *= corr;
            acc1[r] *= corr;
            const int m = r + (half << 3);
            pbuf[m * 32 + lm]      = (__bf16)p0;
            pbuf[m * 32 + 16 + lm] = (__bf16)p1;
        }
        asm volatile("s_wait_dscnt 0" ::: "memory");

        const __bf16* prow = pbuf + lm * 32;
        const v16bf ap = ld16(prow + half * 8, prow + 16 + half * 8);   // ds_load_b128 x2
        const __bf16* vr0 = vb + (size_t)lm * HW + kt + half * 8;
        const __bf16* vr1 = vb + (size_t)(16 + lm) * HW + kt + half * 8;
        const v16bf bv0 = ld16(vr0, vr0 + 16);
        const v16bf bv1 = ld16(vr1, vr1 + 16);
        acc0 = __builtin_amdgcn_wmma_f32_16x16x32_bf16(false, ap, false, bv0, (short)0, acc0, false, false);
        acc1 = __builtin_amdgcn_wmma_f32_16x16x32_bf16(false, ap, false, bv1, (short)0, acc1, false, false);
    }

    __bf16* avb = avT + (size_t)b * HW * QC;
#pragma unroll
    for (int r = 0; r < 8; ++r) {
        const int m = r + (half << 3);
        const float inv = 1.0f / lrow[r];
        avb[(size_t)(qbase + m) * QC + lm]      = (__bf16)(acc0[r] * inv);
        avb[(size_t)(qbase + m) * QC + 16 + lm] = (__bf16)(acc1[r] * inv);
    }
}

// ---------------------------------------------------------------------------
// Elementwise / layout kernels (pixel-major activations)
// ---------------------------------------------------------------------------
__global__ void cvt_bf16_kernel(const float* __restrict__ s, __bf16* __restrict__ d, int n) {
    const int i = blockIdx.x * blockDim.x + threadIdx.x;
    if (i < n) d[i] = (__bf16)s[i];
}

// x f32 CHW -> xT bf16 [B][HW][256]
__global__ void transpose_x_kernel(const float* __restrict__ x, __bf16* __restrict__ xT) {
    const int idx = blockIdx.x * blockDim.x + threadIdx.x;   // BB*HW
    if (idx >= BB * HW) return;
    const int b = idx >> 12, pos = idx & 4095;
    const float* xp = x + (size_t)b * IN_CH * HW + pos;
    __bf16* op = xT + (size_t)idx * IN_CH;
    for (int c = 0; c < IN_CH; ++c) op[c] = (__bf16)xp[(size_t)c * HW];
}

// skip f32 [B][64][32][32] -> suT f32 [B][HW][64], bilinear half-pixel x2
__global__ void upsample_kernel(const float* __restrict__ skip, float* __restrict__ suT) {
    const int idx = blockIdx.x * blockDim.x + threadIdx.x;   // BB*HW*64, c fastest
    if (idx >= BB * HW * SKIP_CH) return;
    const int c = idx & 63, p = idx >> 6;
    const int pos = p & 4095, b = p >> 12;
    const int y = pos >> 6, x = pos & 63;
    const float sy = y * 0.5f - 0.25f, sx = x * 0.5f - 0.25f;
    const float y0f = floorf(sy), x0f = floorf(sx);
    const int y0 = (int)y0f, x0 = (int)x0f;
    const float wy = sy - y0f, wx = sx - x0f;
    const int y0c = max(y0, 0), y1c = min(y0 + 1, 31);
    const int x0c = max(x0, 0), x1c = min(x0 + 1, 31);
    const float* sp = skip + (size_t)(b * SKIP_CH + c) * 1024;
    suT[idx] = (1.f - wy) * ((1.f - wx) * sp[y0c * 32 + x0c] + wx * sp[y0c * 32 + x1c])
             +        wy  * ((1.f - wx) * sp[y1c * 32 + x0c] + wx * sp[y1c * 32 + x1c]);
}

__global__ void mean_kernel(const float* __restrict__ suT, float* __restrict__ mb) {
    const int idx = blockIdx.x * blockDim.x + threadIdx.x;   // BB*HW
    if (idx >= BB * HW) return;
    const float* sp = suT + (size_t)idx * SKIP_CH;
    float s = 0.f;
    for (int c = 0; c < SKIP_CH; ++c) s += sp[c];
    mb[idx] = s * (1.0f / SKIP_CH);
}

__global__ void edge_kernel(const float* __restrict__ mb, const float* __restrict__ ew,
                            float* __restrict__ edgeT) {
    const int idx = blockIdx.x * blockDim.x + threadIdx.x;   // BB*HW*64, oc fastest
    if (idx >= BB * HW * SKIP_CH) return;
    const int oc = idx & 63, p = idx >> 6;
    const int pos = p & 4095, b = p >> 12;
    const int y = pos >> 6, x = pos & 63;
    float acc = 0.f;
#pragma unroll
    for (int ky = 0; ky < 3; ++ky)
#pragma unroll
        for (int kx = 0; kx < 3; ++kx) {
            const int yy = y + ky - 1, xx = x + kx - 1;
            if (yy >= 0 && yy < HH && xx >= 0 && xx < WW)
                acc += mb[b * HW + yy * WW + xx] * ew[oc * 9 + ky * 3 + kx];
        }
    edgeT[idx] = acc;
}

__global__ void thick_kernel(const float* __restrict__ suT, const float* __restrict__ tw,
                             const float* __restrict__ tb, float* __restrict__ thick) {
    const int idx = blockIdx.x * blockDim.x + threadIdx.x;   // BB*HW
    if (idx >= BB * HW) return;
    const float* sp = suT + (size_t)idx * SKIP_CH;
    float s = tb[0];
    for (int c = 0; c < SKIP_CH; ++c) s += sp[c] * tw[c];
    thick[idx] = 1.0f / (1.0f + __expf(-s));
}

__global__ void off_kernel(const float* __restrict__ edgeT, const float* __restrict__ ow,
                           const float* __restrict__ ob, const float* __restrict__ thick,
                           float* __restrict__ off) {
    const int idx = blockIdx.x * blockDim.x + threadIdx.x;   // BB*18*HW
    if (idx >= BB * 18 * HW) return;
    const int pos = idx & 4095, boc = idx >> 12;
    const int oc = boc % 18, b = boc / 18;
    const int y = pos >> 6, x = pos & 63;
    float acc = ob[oc];
#pragma unroll
    for (int ky = 0; ky < 3; ++ky)
#pragma unroll
        for (int kx = 0; kx < 3; ++kx) {
            const int yy = y + ky - 1, xx = x + kx - 1;
            if (yy >= 0 && yy < HH && xx >= 0 && xx < WW) {
                const float* ep = edgeT + (size_t)(b * HW + yy * WW + xx) * SKIP_CH;
                const float* wp = ow + (oc * SKIP_CH) * 9 + ky * 3 + kx;
                for (int c = 0; c < SKIP_CH; ++c) acc += ep[c] * wp[c * 9];
            }
        }
    off[(size_t)(b * 18 + oc) * HW + pos] = acc * (1.0f + 16.0f * thick[b * HW + pos]);
}

// DCNv1 im2col-with-bilinear-gather -> valsT bf16 [B][HW][576] (col = c*9+k)
__global__ void deform_gather_kernel(const float* __restrict__ suT, const float* __restrict__ off,
                                     __bf16* __restrict__ valsT) {
    const int idx = blockIdx.x * blockDim.x + threadIdx.x;   // BB*9*HW
    if (idx >= BB * 9 * HW) return;
    const int pos = idx & 4095, bk = idx >> 12;
    const int k = bk % 9, b = bk / 9;
    const int y = pos >> 6, x = pos & 63;
    const float dy = off[(size_t)(b * 18 + 2 * k) * HW + pos];
    const float dx = off[(size_t)(b * 18 + 2 * k + 1) * HW + pos];
    const float py = (float)(y + (k / 3) - 1) + dy;
    const float px = (float)(x + (k % 3) - 1) + dx;
    const float y0f = floorf(py), x0f = floorf(px);
    const int y0 = (int)y0f, x0 = (int)x0f;
    const float wy = py - y0f, wx = px - x0f;
    const int y1 = y0 + 1, x1 = x0 + 1;
    const float m00 = (y0 >= 0 && y0 < HH && x0 >= 0 && x0 < WW) ? 1.f : 0.f;
    const float m01 = (y0 >= 0 && y0 < HH && x1 >= 0 && x1 < WW) ? 1.f : 0.f;
    const float m10 = (y1 >= 0 && y1 < HH && x0 >= 0 && x0 < WW) ? 1.f : 0.f;
    const float m11 = (y1 >= 0 && y1 < HH && x1 >= 0 && x1 < WW) ? 1.f : 0.f;
    const int y0c = min(max(y0, 0), HH - 1), y1c = min(max(y1, 0), HH - 1);
    const int x0c = min(max(x0, 0), WW - 1), x1c = min(max(x1, 0), WW - 1);
    const float w00 = (1.f - wy) * (1.f - wx) * m00;
    const float w01 = (1.f - wy) * wx * m01;
    const float w10 = wy * (1.f - wx) * m10;
    const float w11 = wy * wx * m11;
    const float* r00 = suT + (size_t)(b * HW + y0c * WW + x0c) * SKIP_CH;
    const float* r01 = suT + (size_t)(b * HW + y0c * WW + x1c) * SKIP_CH;
    const float* r10 = suT + (size_t)(b * HW + y1c * WW + x0c) * SKIP_CH;
    const float* r11 = suT + (size_t)(b * HW + y1c * WW + x1c) * SKIP_CH;
    __bf16* op = valsT + (size_t)(b * HW + pos) * 576 + k;
    for (int c = 0; c < SKIP_CH; ++c)
        op[c * 9] = (__bf16)(w00 * r00[c] + w01 * r01[c] + w10 * r10[c] + w11 * r11[c]);
}

// LayerNorm over 64 channels (contiguous rows) -> bf16
__global__ void ln_kernel(const float* __restrict__ eT, const float* __restrict__ g,
                          const float* __restrict__ bt, __bf16* __restrict__ hnT) {
    const int idx = blockIdx.x * blockDim.x + threadIdx.x;   // BB*HW
    if (idx >= BB * HW) return;
    const float* ep = eT + (size_t)idx * EMBED;
    float s = 0.f, s2 = 0.f;
    for (int c = 0; c < EMBED; ++c) { const float v = ep[c]; s += v; s2 += v * v; }
    const float mu = s * (1.0f / EMBED);
    const float var = s2 * (1.0f / EMBED) - mu * mu;
    const float inv = rsqrtf(var + 1e-5f);
    __bf16* op = hnT + (size_t)idx * EMBED;
    for (int c = 0; c < EMBED; ++c) op[c] = (__bf16)((ep[c] - mu) * inv * g[c] + bt[c]);
}

// depthwise 3x3 + bias + exact GELU, pixel-major bf16
__global__ void dwconv_kernel(const __bf16* __restrict__ h1T, const float* __restrict__ dww,
                              const float* __restrict__ dwb, __bf16* __restrict__ h2T) {
    const int idx = blockIdx.x * blockDim.x + threadIdx.x;   // BB*HW*256, c fastest
    if (idx >= BB * HW * 256) return;
    const int c = idx & 255, p = idx >> 8;
    const int pos = p & 4095, b = p >> 12;
    const int y = pos >> 6, x = pos & 63;
    float acc = dwb[c];
#pragma unroll
    for (int ky = 0; ky < 3; ++ky)
#pragma unroll
        for (int kx = 0; kx < 3; ++kx) {
            const int yy = y + ky - 1, xx = x + kx - 1;
            if (yy >= 0 && yy < HH && xx >= 0 && xx < WW)
                acc += (float)h1T[(size_t)(b * HW + yy * WW + xx) * 256 + c] * dww[c * 9 + ky * 3 + kx];
        }
    h2T[idx] = (__bf16)gelu_exact(acc);
}

// ---------------------------------------------------------------------------
extern "C" void kernel_launch(void* const* d_in, const int* in_sizes, int n_in,
                              void* d_out, int out_size, void* d_ws, size_t ws_size,
                              hipStream_t stream) {
    const float* x       = (const float*)d_in[0];
    const float* skip    = (const float*)d_in[1];
    const float* qkv_w   = (const float*)d_in[2];
    const float* qkv_b   = (const float*)d_in[3];
    const float* proj_w  = (const float*)d_in[4];
    const float* proj_b  = (const float*)d_in[5];
    const float* edge_w  = (const float*)d_in[6];
    const float* thick_w = (const float*)d_in[7];
    const float* thick_b = (const float*)d_in[8];
    const float* off_w   = (const float*)d_in[9];
    const float* off_b   = (const float*)d_in[10];
    const float* dfm_w   = (const float*)d_in[11];
    const float* dfm_b   = (const float*)d_in[12];
    const float* ln_g    = (const float*)d_in[13];
    const float* ln_b    = (const float*)d_in[14];
    const float* mlp1_w  = (const float*)d_in[15];
    const float* mlp1_b  = (const float*)d_in[16];
    const float* dw_w    = (const float*)d_in[17];
    const float* dw_b    = (const float*)d_in[18];
    const float* mlp2_w  = (const float*)d_in[19];
    const float* mlp2_b  = (const float*)d_in[20];
    const float* down_w  = (const float*)d_in[21];
    const float* down_b  = (const float*)d_in[22];
    const float* up_w    = (const float*)d_in[23];
    const float* up_b    = (const float*)d_in[24];
    float* out = (float*)d_out;

    // ---- workspace carve-out (256B aligned) ----
    char* base = (char*)d_ws;
    size_t o = 0;
    auto alloc = [&](size_t bytes) -> void* {
        o = (o + 255) & ~(size_t)255;
        void* p = base + o;
        o += bytes;
        return p;
    };
    float*  suT   = (float*) alloc((size_t)BB * HW * SKIP_CH * 4);
    float*  mb    = (float*) alloc((size_t)BB * HW * 4);
    float*  edgeT = (float*) alloc((size_t)BB * HW * SKIP_CH * 4);
    float*  thick = (float*) alloc((size_t)BB * HW * 4);
    float*  offb  = (float*) alloc((size_t)BB * 18 * HW * 4);
    __bf16* xT    = (__bf16*)alloc((size_t)BB * HW * IN_CH * 2);
    __bf16* qkT   = (__bf16*)alloc((size_t)BB * HW * 64 * 2);
    __bf16* vbf   = (__bf16*)alloc((size_t)BB * QC * HW * 2);
    __bf16* avT   = (__bf16*)alloc((size_t)BB * HW * QC * 2);
    __bf16* catT  = (__bf16*)alloc((size_t)BB * HW * 320 * 2);
    __bf16* valsT = (__bf16*)alloc((size_t)BB * HW * 576 * 2);   // reused for h1T/h2T
    float*  eT    = (float*) alloc((size_t)BB * HW * EMBED * 4);
    __bf16* hnT   = (__bf16*)alloc((size_t)BB * HW * EMBED * 2);
    __bf16* tT    = (__bf16*)alloc((size_t)BB * HW * EMBED * 2);
    __bf16* qkv_wb  = (__bf16*)alloc(96 * 256 * 2);
    __bf16* proj_wb = (__bf16*)alloc(256 * 32 * 2);
    __bf16* dfm_wb  = (__bf16*)alloc(64 * 576 * 2);
    __bf16* down_wb = (__bf16*)alloc(64 * 320 * 2);
    __bf16* mlp1_wb = (__bf16*)alloc(256 * 64 * 2);
    __bf16* mlp2_wb = (__bf16*)alloc(64 * 256 * 2);
    __bf16* up_wb   = (__bf16*)alloc(256 * 64 * 2);
    __bf16* h1T = valsT;                                 // valsT dead after deform GEMM
    __bf16* h2T = valsT + (size_t)BB * HW * 256;

    const int EW = 256;
#define GRID1(n) (((n) + EW - 1) / EW), EW, 0, stream

    // ---- weight + input conversions (bf16, tiny) ----
    cvt_bf16_kernel<<<GRID1(96 * 256)>>>(qkv_w, qkv_wb, 96 * 256);
    cvt_bf16_kernel<<<GRID1(256 * 32)>>>(proj_w, proj_wb, 256 * 32);
    cvt_bf16_kernel<<<GRID1(64 * 576)>>>(dfm_w, dfm_wb, 64 * 576);
    cvt_bf16_kernel<<<GRID1(64 * 320)>>>(down_w, down_wb, 64 * 320);
    cvt_bf16_kernel<<<GRID1(256 * 64)>>>(mlp1_w, mlp1_wb, 256 * 64);
    cvt_bf16_kernel<<<GRID1(64 * 256)>>>(mlp2_w, mlp2_wb, 64 * 256);
    cvt_bf16_kernel<<<GRID1(256 * 64)>>>(up_w, up_wb, 256 * 64);
    transpose_x_kernel<<<GRID1(BB * HW)>>>(x, xT);

    // ---- skip branch prologue ----
    upsample_kernel<<<GRID1(BB * HW * SKIP_CH)>>>(skip, suT);
    mean_kernel<<<GRID1(BB * HW)>>>(suT, mb);
    edge_kernel<<<GRID1(BB * HW * SKIP_CH)>>>(mb, edge_w, edgeT);
    thick_kernel<<<GRID1(BB * HW)>>>(suT, thick_w, thick_b, thick);
    off_kernel<<<GRID1(BB * 18 * HW)>>>(edgeT, off_w, off_b, thick, offb);

    // ---- attention path [WMMA] ----
    qkv_gemm_kernel<<<(BB * 6 * 256) / 4, 128, 0, stream>>>(qkv_wb, xT, qkv_b, qkT, vbf);
    attn_kernel<<<(BB * 256) / 4, 128, 0, stream>>>(qkT, vbf, avT);
    // x2 = proj@av + x -> catT columns [0,256)
    tgemm_kernel<256, 32, 32, 1, 0, 0, 320, 0>
        <<<(BB * 16 * 256) / 4, 128, 0, stream>>>(proj_wb, avT, proj_b, xT, catT);

    // ---- deformable path [WMMA] -> catT columns [256,320) ----
    deform_gather_kernel<<<GRID1(BB * 9 * HW)>>>(suT, offb, valsT);
    tgemm_kernel<64, 576, 576, 2, 0, 0, 320, 256>
        <<<(BB * 4 * 256) / 4, 128, 0, stream>>>(dfm_wb, valsT, dfm_b, edgeT, catT);

    // ---- down (fused concat via catT) [WMMA] ----
    tgemm_kernel<64, 320, 320, 0, 0, 1, 64, 0>
        <<<(BB * 4 * 256) / 4, 128, 0, stream>>>(down_wb, catT, down_b, nullptr, eT);

    // ---- token mixer ----
    ln_kernel<<<GRID1(BB * HW)>>>(eT, ln_g, ln_b, hnT);
    tgemm_kernel<256, 64, 64, 0, 1, 0, 256, 0>
        <<<(BB * 16 * 256) / 4, 128, 0, stream>>>(mlp1_wb, hnT, mlp1_b, nullptr, h1T);
    dwconv_kernel<<<GRID1(BB * HW * 256)>>>(h1T, dw_w, dw_b, h2T);
    tgemm_kernel<64, 256, 256, 2, 0, 0, 64, 0>
        <<<(BB * 4 * 256) / 4, 128, 0, stream>>>(mlp2_wb, h2T, mlp2_b, eT, tT);
    // ---- up-proj -> f32 CHW output ----
    tgemm_kernel<256, 64, 64, 0, 0, 2, 0, 0>
        <<<(BB * 16 * 256) / 4, 128, 0, stream>>>(up_wb, tT, up_b, nullptr, out);
#undef GRID1
}
